// LongShortTermMemoryNetworks_15470472200298
// MI455X (gfx1250) — matrix operand order repaired and checked by reading, not compiled, not run on hardware
//
#include <hip/hip_runtime.h>

// ---------------------------------------------------------------------------
// Bidirectional LSTM for MI455X (gfx1250, wave32, WMMA).
//
//   1) prep: embeddings+feats -> bf16 A (51200 x 320), Wih/Whh/Wfc -> bf16,
//      fused bias.
//   2) xproj: A @ Wih^T + bias (wmma bf16), M tiled by (batch-tile, t),
//      output stored in WMMA C-fragment-permuted layout
//      [dir][bt][t][ntile][lane][8] so the recurrence reads it as b128.
//   3) lstm: grid (16,2), 16 batch rows/WG, 200 steps, ONE barrier per step.
//      Zero global weight traffic in the loop: lower K-half of Whh in LDS
//      (256KB staged once, padded rows -> conflict-free ds_load_b128),
//      upper K-half in persistent per-wave register B-fragments (256 VGPRs).
//      WMMA chains start from C=0; the streamed x_proj tile is folded in at
//      the elementwise phase, so its global-load latency hides behind the
//      64-WMMA chain. h double-buffered bf16 in LDS; c in registers;
//      h exported to global via coalesced b128 copies.
//   4) fc: rnn(bf16 51200x512) @ Wfc^T(48-pad) -> fp32 51200x39.
// ---------------------------------------------------------------------------

#define V_   50000
#define D_   300
#define H_   256
#define B_   256
#define L_   200
#define NL_  39
#define G_   1024        // 4*H
#define DIN_ 303
#define KP_  320         // DIN padded to multiple of 32
#define M_   (B_ * L_)   // 51200
#define NFC_ 48          // 39 padded to 3 n-tiles

#define WLD_  136        // LDS Whh row pitch (128 + 8 pad), u16 elements
#define HLD_  264        // LDS h row pitch (256 + 8 pad), u16 elements

typedef __bf16 bf16_t;
typedef __attribute__((ext_vector_type(16))) __bf16 v16bf;
typedef __attribute__((ext_vector_type(8)))  __bf16 v8bf;
typedef __attribute__((ext_vector_type(8)))  float  v8f;
typedef __attribute__((ext_vector_type(4)))  float  v4f;

union V16U { v16bf v; v8bf h[2]; };
union V8FU { v8f  v; v4f h[2]; };

__device__ __forceinline__ unsigned short f2bf(float f) {
    unsigned int u = __float_as_uint(f);
    u += 0x7FFFu + ((u >> 16) & 1u);          // round-to-nearest-even
    return (unsigned short)(u >> 16);
}

__device__ __forceinline__ v8f v8f_zero() {
    v8f z;
#pragma unroll
    for (int i = 0; i < 8; ++i) z[i] = 0.0f;
    return z;
}

__device__ __forceinline__ float sigmoidf_(float x) {
    return 1.0f / (1.0f + __expf(-x));
}

__device__ __forceinline__ v8f wmma_bf16(v16bf a, v16bf b, v8f c) {
    return __builtin_amdgcn_wmma_f32_16x16x32_bf16(
        false, a, false, b, (short)0, c, false, false);
}

// A fragment (16x32 bf16). ISA 7.12.2: lanes 0-15 M=lane, K chunks
// {k0+ph*8..+8} and {k0+16+ph*8..+8} with ph = lane>>4. 16B loads.
__device__ __forceinline__ v16bf load_a_frag(const unsigned short* base, int lda,
                                             int m0, int k0, int lane) {
    const int r  = lane & 15;
    const int ph = lane >> 4;
    const unsigned short* p = base + (size_t)(m0 + r) * lda + k0 + ph * 8;
    V16U u;
    u.h[0] = *(const v8bf*)(p);
    u.h[1] = *(const v8bf*)(p + 16);
    return u.v;
}

// B fragment (32x16 as KxN) from row-major W[N][K]: lane holds column
// N = n0+(lane&15), K chunks {k0+ph*8..+8} and {k0+16+ph*8..+8}, ph = lane>>4.
__device__ __forceinline__ v16bf load_b_frag(const unsigned short* base, int ldb,
                                             int n0, int k0, int lane) {
    const int n  = lane & 15;
    const int ph = lane >> 4;
    const unsigned short* p = base + (size_t)(n0 + n) * ldb + k0 + ph * 8;
    V16U u;
    u.h[0] = *(const v8bf*)(p);
    u.h[1] = *(const v8bf*)(p + 16);
    return u.v;
}

// ---------------------------------------------------------------------------
// Prep kernels
// ---------------------------------------------------------------------------

__global__ void prep_inputs_kernel(const int* __restrict__ x,
                                   const float* __restrict__ pre,
                                   unsigned short* __restrict__ Abf) {
    size_t idx = (size_t)blockIdx.x * blockDim.x + threadIdx.x;
    if (idx >= (size_t)M_ * KP_) return;
    int col = (int)(idx % KP_);
    int row = (int)(idx / KP_);
    int b = row / L_, t = row % L_;
    float v = 0.0f;
    if (col < D_) {
        int tok = x[((size_t)b * 4 + 0) * L_ + t];
        v = pre[(size_t)tok * D_ + col];
    } else if (col < DIN_) {
        v = (float)x[((size_t)b * 4 + 1 + (col - D_)) * L_ + t];
    }
    Abf[idx] = f2bf(v);
}

__global__ void prep_wih_kernel(const float* __restrict__ wf,
                                const float* __restrict__ wb,
                                unsigned short* __restrict__ o) {
    int idx = blockIdx.x * blockDim.x + threadIdx.x;
    if (idx >= 2 * G_ * KP_) return;
    int col = idx % KP_;
    int r   = (idx / KP_) % G_;
    int dir = idx / (KP_ * G_);
    const float* w = dir ? wb : wf;
    float v = (col < DIN_) ? w[(size_t)r * DIN_ + col] : 0.0f;
    o[idx] = f2bf(v);
}

__global__ void prep_whh_kernel(const float* __restrict__ wf,
                                const float* __restrict__ wb,
                                unsigned short* __restrict__ o) {
    int idx = blockIdx.x * blockDim.x + threadIdx.x;
    if (idx >= 2 * G_ * H_) return;
    int col = idx % H_;
    int r   = (idx / H_) % G_;
    int dir = idx / (H_ * G_);
    const float* w = dir ? wb : wf;
    o[idx] = f2bf(w[(size_t)r * H_ + col]);
}

__global__ void prep_wfc_kernel(const float* __restrict__ wfc,
                                unsigned short* __restrict__ o) {
    int idx = blockIdx.x * blockDim.x + threadIdx.x;
    if (idx >= NFC_ * 2 * H_) return;
    int col = idx % (2 * H_);
    int r   = idx / (2 * H_);
    float v = (r < NL_) ? wfc[(size_t)r * 2 * H_ + col] : 0.0f;
    o[idx] = f2bf(v);
}

__global__ void prep_bias_kernel(const float* __restrict__ bihf,
                                 const float* __restrict__ bhhf,
                                 const float* __restrict__ bihb,
                                 const float* __restrict__ bhhb,
                                 float* __restrict__ o) {
    int idx = blockIdx.x * blockDim.x + threadIdx.x;
    if (idx >= 2 * G_) return;
    int g = idx & (G_ - 1);
    o[idx] = (idx >= G_) ? (bihb[g] + bhhb[g]) : (bihf[g] + bhhf[g]);
}

// ---------------------------------------------------------------------------
// x_proj GEMM, M tiled by (batch, t). grid = (L_, 16, 2):
//   blockIdx.y = mhalf(2) * nchunk(8). WG = 8 waves: 4 M-slabs(32 rows b) x
//   2 N-slabs(64). Output in C-fragment-permuted layout:
//   xpP[(((dir*16+bt)*L + t)*64 + ntile)*256 + lane*8 + r]
// ---------------------------------------------------------------------------
__global__ __launch_bounds__(256)
void xproj_kernel(const unsigned short* __restrict__ Abf,
                  const unsigned short* __restrict__ Wihb,
                  const float* __restrict__ biasSum,
                  float* __restrict__ xpP) {
    const int t      = blockIdx.x;
    const int mhalf  = blockIdx.y >> 3;
    const int nchunk = blockIdx.y & 7;
    const int dir    = blockIdx.z;
    const int wave   = threadIdx.x >> 5;
    const int lane   = threadIdx.x & 31;
    const int mslab  = wave >> 1;
    const int nslab  = wave & 1;
    const int b0     = mhalf * 128 + mslab * 32;      // batch row base (32 rows)
    const int n0w    = nchunk * 128 + nslab * 64;
    const unsigned short* W = Wihb + (size_t)dir * G_ * KP_;
    const unsigned short* A = Abf + (size_t)t * KP_;  // row index = b, stride L*KP

    v8f acc[2][4];
#pragma unroll
    for (int mt = 0; mt < 2; ++mt)
#pragma unroll
        for (int nt = 0; nt < 4; ++nt) acc[mt][nt] = v8f_zero();

    for (int kt = 0; kt < KP_ / 32; ++kt) {
        v16bf a0 = load_a_frag(A, L_ * KP_, b0,      kt * 32, lane);
        v16bf a1 = load_a_frag(A, L_ * KP_, b0 + 16, kt * 32, lane);
#pragma unroll
        for (int nt = 0; nt < 4; ++nt) {
            v16bf b = load_b_frag(W, KP_, n0w + nt * 16, kt * 32, lane);
            acc[0][nt] = wmma_bf16(a0, b, acc[0][nt]);
            acc[1][nt] = wmma_bf16(a1, b, acc[1][nt]);
        }
    }

    const int rl = lane & 15;
#pragma unroll
    for (int nt = 0; nt < 4; ++nt) {
        int ntile = (n0w >> 4) + nt;
        float bias = biasSum[dir * G_ + ntile * 16 + rl];
#pragma unroll
        for (int mt = 0; mt < 2; ++mt) {
            int bt = (b0 >> 4) + mt;
            V8FU u;
            u.v = acc[mt][nt];
#pragma unroll
            for (int r = 0; r < 8; ++r) u.v[r] += bias;
            float* p = xpP +
                ((((size_t)(dir * 16 + bt) * L_ + t) * 64 + ntile) * 256) + lane * 8;
            *(v4f*)(p)     = u.h[0];
            *(v4f*)(p + 4) = u.h[1];
        }
    }
}

// ---------------------------------------------------------------------------
// LSTM recurrence. grid = (B/16, 2 dirs), block = 256 (8 waves).
// Wave w owns hidden units [w*32, w*32+32): tiles at gate*256 + w*32 + {0,16}.
// Whh K=0..127 in LDS, K=128..255 in persistent registers. h double-buffered
// in LDS -> single barrier per timestep. x_proj added at the elementwise
// phase so its load latency hides behind the WMMA chain.
// ---------------------------------------------------------------------------
__global__ __launch_bounds__(256)
void lstm_kernel(const float* __restrict__ xpP,          // fragment-permuted
                 const unsigned short* __restrict__ whhb, // [2][G_][H_]
                 unsigned short* __restrict__ rnn) {      // [M_][2H_] bf16
    __shared__ __align__(16) unsigned short whh_lds[G_ * WLD_]; // 272 KB
    __shared__ __align__(16) unsigned short hbf[2][16 * HLD_];  // 2 x 8.25 KB

    const int dir  = blockIdx.y;
    const int b0   = blockIdx.x * 16;
    const int bt   = blockIdx.x;
    const int tid  = threadIdx.x;
    const int lane = tid & 31;
    const int wave = tid >> 5;
    const int j0   = wave * 32;
    const int rl   = lane & 15;
    const int ph   = lane >> 4;

    const unsigned short* whh = whhb + (size_t)dir * G_ * H_;
    const float* xbase = xpP + (((size_t)(dir * 16 + bt) * L_) * 64) * 256;

    // ---- stage lower K-half of Whh (1024 x 128 bf16) into padded LDS ----
    for (int idx = tid; idx < G_ * 64; idx += 256) {   // dword copies
        int row = idx >> 6;
        int cd  = idx & 63;
        unsigned int v = *(const unsigned int*)(whh + (size_t)row * H_ + cd * 2);
        *(unsigned int*)(&whh_lds[row * WLD_ + cd * 2]) = v;
    }
    // ---- zero h buffer 0 ----
    for (int i = tid; i < 16 * HLD_ / 2; i += 256)
        *(unsigned int*)(&hbf[0][i * 2]) = 0u;

    // ---- persistent register B-fragments: K = 128..255 ----
    v16bf breg[8][4];
#pragma unroll
    for (int g = 0; g < 4; ++g)
#pragma unroll
        for (int hh = 0; hh < 2; ++hh)
#pragma unroll
            for (int kt = 0; kt < 4; ++kt)
                breg[g * 2 + hh][kt] =
                    load_b_frag(whh, H_, g * H_ + j0 + hh * 16,
                                128 + kt * 32, lane);

    float c[2][8];
#pragma unroll
    for (int hh = 0; hh < 2; ++hh)
#pragma unroll
        for (int r = 0; r < 8; ++r) c[hh][r] = 0.0f;
    __syncthreads();

    const int crow = tid >> 4;           // h-export: row this thread copies
    const int cseg = tid & 15;           // 16-unit segment within the row

    for (int step = 0; step < L_; ++step) {
        const int t   = dir ? (L_ - 1 - step) : step;
        const int cur = step & 1;
        const int nxt = cur ^ 1;
        const unsigned short* hsrc = hbf[cur];
        unsigned short*       hdst = hbf[nxt];

        // stream x_proj tile (issued now, consumed only after the WMMA chain)
        v8f xv[4][2];
        const float* xt = xbase + (size_t)t * 64 * 256;
#pragma unroll
        for (int g = 0; g < 4; ++g)
#pragma unroll
            for (int hh = 0; hh < 2; ++hh) {
                const float* p = xt + (g * 16 + wave * 2 + hh) * 256 + lane * 8;
                V8FU u;
                u.h[0] = __builtin_nontemporal_load((const v4f*)(p));
                u.h[1] = __builtin_nontemporal_load((const v4f*)(p + 4));
                xv[g][hh] = u.v;
            }

        // A fragments of h from LDS
        v16bf afrag[8];
#pragma unroll
        for (int kt = 0; kt < 8; ++kt)
            afrag[kt] = load_a_frag(hsrc, HLD_, 0, kt * 32, lane);

        // gates = h @ Whh^T, accumulated from C = 0
        v8f acc[4][2];
#pragma unroll
        for (int g = 0; g < 4; ++g)
#pragma unroll
            for (int hh = 0; hh < 2; ++hh)
                acc[g][hh] = v8f_zero();

        // K = 128..255: B in registers, zero memory traffic
#pragma unroll
        for (int kt = 0; kt < 4; ++kt)
#pragma unroll
            for (int g = 0; g < 4; ++g)
#pragma unroll
                for (int hh = 0; hh < 2; ++hh)
                    acc[g][hh] = wmma_bf16(afrag[4 + kt],
                                           breg[g * 2 + hh][kt], acc[g][hh]);

        // K = 0..127: B from LDS
#pragma unroll
        for (int kt = 0; kt < 4; ++kt)
#pragma unroll
            for (int g = 0; g < 4; ++g)
#pragma unroll
                for (int hh = 0; hh < 2; ++hh) {
                    int n0 = g * H_ + j0 + hh * 16;
                    v16bf b = load_b_frag(whh_lds, WLD_, n0, kt * 32, lane);
                    acc[g][hh] = wmma_bf16(afrag[kt], b, acc[g][hh]);
                }

        // nonlinearity + state update; write h_t into the OTHER buffer
#pragma unroll
        for (int hh = 0; hh < 2; ++hh) {
            int unit = j0 + hh * 16 + rl;
#pragma unroll
            for (int r = 0; r < 8; ++r) {
                int row = r + 8 * ph;
                float iv = sigmoidf_(acc[0][hh][r] + xv[0][hh][r]);
                float fv = sigmoidf_(acc[1][hh][r] + xv[1][hh][r]);
                float gv = tanhf(acc[2][hh][r] + xv[2][hh][r]);
                float ov = sigmoidf_(acc[3][hh][r] + xv[3][hh][r]);
                float cn = fv * c[hh][r] + iv * gv;
                c[hh][r] = cn;
                float h = ov * tanhf(cn);
                hdst[row * HLD_ + unit] = f2bf(h);
            }
        }
        __syncthreads();   // h_t complete; visible to next step + export

        // export h_t to global: coalesced b128 copies (32B per thread)
        {
            const v8bf* s = (const v8bf*)(hdst + crow * HLD_ + cseg * 16);
            v8bf* d = (v8bf*)(rnn +
                (((size_t)(b0 + crow) * L_ + t) * (2 * H_)) + dir * H_ + cseg * 16);
            d[0] = s[0];
            d[1] = s[1];
        }
    }
}

// ---------------------------------------------------------------------------
// Final FC: out[m][n<39] = rnn[m][:] . Wfc[n][:] + bfc[n]
// ---------------------------------------------------------------------------
__global__ __launch_bounds__(256)
void fc_kernel(const unsigned short* __restrict__ rnn,   // [M_][512] bf16
               const unsigned short* __restrict__ Wfcb,  // [48][512] bf16
               const float* __restrict__ bfc,
               float* __restrict__ out) {                // [M_][39]
    const int wave = threadIdx.x >> 5;
    const int lane = threadIdx.x & 31;
    const int m0   = blockIdx.x * 256 + wave * 32;

    v8f acc[2][3];
#pragma unroll
    for (int mt = 0; mt < 2; ++mt)
#pragma unroll
        for (int nt = 0; nt < 3; ++nt) acc[mt][nt] = v8f_zero();

    for (int kt = 0; kt < (2 * H_) / 32; ++kt) {
        v16bf a0 = load_a_frag(rnn, 2 * H_, m0,      kt * 32, lane);
        v16bf a1 = load_a_frag(rnn, 2 * H_, m0 + 16, kt * 32, lane);
#pragma unroll
        for (int nt = 0; nt < 3; ++nt) {
            v16bf b = load_b_frag(Wfcb, 2 * H_, nt * 16, kt * 32, lane);
            acc[0][nt] = wmma_bf16(a0, b, acc[0][nt]);
            acc[1][nt] = wmma_bf16(a1, b, acc[1][nt]);
        }
    }

    const int rl = lane & 15;
    const int ph = lane >> 4;
#pragma unroll
    for (int nt = 0; nt < 3; ++nt) {
        int n = nt * 16 + rl;
        if (n < NL_) {
            float bias = bfc[n];
#pragma unroll
            for (int mt = 0; mt < 2; ++mt)
#pragma unroll
                for (int r = 0; r < 8; ++r) {
                    int m = m0 + mt * 16 + r + 8 * ph;
                    out[(size_t)m * NL_ + n] = acc[mt][nt][r] + bias;
                }
        }
    }
}

// ---------------------------------------------------------------------------
// Launch
// ---------------------------------------------------------------------------
extern "C" void kernel_launch(void* const* d_in, const int* in_sizes, int n_in,
                              void* d_out, int out_size, void* d_ws, size_t ws_size,
                              hipStream_t stream) {
    const int*   x     = (const int*)  d_in[0];
    const float* pre   = (const float*)d_in[1];
    const float* wih_f = (const float*)d_in[2];
    const float* whh_f = (const float*)d_in[3];
    const float* bih_f = (const float*)d_in[4];
    const float* bhh_f = (const float*)d_in[5];
    const float* wih_b = (const float*)d_in[6];
    const float* whh_b = (const float*)d_in[7];
    const float* bih_b = (const float*)d_in[8];
    const float* bhh_b = (const float*)d_in[9];
    const float* wfc   = (const float*)d_in[10];
    const float* bfc   = (const float*)d_in[11];
    float* out = (float*)d_out;

    // Workspace carve-up (256B aligned). Total ~507 MB.
    char* ws = (char*)d_ws;
    size_t off = 0;
    auto take = [&](size_t bytes) {
        char* p = ws + off;
        off = (off + bytes + 255) & ~(size_t)255;
        return p;
    };
    unsigned short* Abf     = (unsigned short*)take((size_t)M_ * KP_ * 2);
    unsigned short* Wihb    = (unsigned short*)take((size_t)2 * G_ * KP_ * 2);
    unsigned short* Whhb    = (unsigned short*)take((size_t)2 * G_ * H_ * 2);
    float*          biasSum = (float*)         take((size_t)2 * G_ * 4);
    unsigned short* Wfcb    = (unsigned short*)take((size_t)NFC_ * 2 * H_ * 2);
    unsigned short* rnn     = (unsigned short*)take((size_t)M_ * 2 * H_ * 2);
    float*          xpP     = (float*)         take((size_t)2 * M_ * G_ * 4);
    (void)ws_size; (void)in_sizes; (void)n_in; (void)out_size;

    // 1) prep
    {
        size_t n = (size_t)M_ * KP_;
        prep_inputs_kernel<<<dim3((unsigned)((n + 255) / 256)), 256, 0, stream>>>(x, pre, Abf);
    }
    prep_wih_kernel<<<(2 * G_ * KP_ + 255) / 256, 256, 0, stream>>>(wih_f, wih_b, Wihb);
    prep_whh_kernel<<<(2 * G_ * H_ + 255) / 256, 256, 0, stream>>>(whh_f, whh_b, Whhb);
    prep_wfc_kernel<<<(NFC_ * 2 * H_ + 255) / 256, 256, 0, stream>>>(wfc, Wfcb);
    prep_bias_kernel<<<(2 * G_ + 255) / 256, 256, 0, stream>>>(bih_f, bhh_f, bih_b, bhh_b, biasSum);

    // 2) input projection GEMM (both directions), fragment-permuted output
    xproj_kernel<<<dim3(L_, 16, 2), 256, 0, stream>>>(Abf, Wihb, biasSum, xpP);

    // 3) recurrence
    lstm_kernel<<<dim3(B_ / 16, 2), 256, 0, stream>>>(xpP, Whhb, rnn);

    // 4) final FC
    fc_kernel<<<M_ / 256, 256, 0, stream>>>(rnn, Wfcb, bfc, out);
}
